// MultiHeadAttention_48533130445634
// MI455X (gfx1250) — matrix operand order repaired and checked by reading, not compiled, not run on hardware
//
#include <hip/hip_runtime.h>

// Problem constants from the reference: shape (4, 16, 2048, 64)
#define BH   64          // B*H
#define SEQ  2048        // N == M
#define DH   64          // head dim
#define ROWS_PER_BLOCK 128
#define MCHUNK 128
#define NTHREADS 256
#define QK_SCALE 0.125f  // (1024/16)^-0.5

typedef __attribute__((ext_vector_type(16))) __bf16 v16bf;
typedef __attribute__((ext_vector_type(8)))  __bf16 v8bf;
typedef __attribute__((ext_vector_type(8)))  float  v8f;

__device__ __forceinline__ unsigned short f2bf_rne(float f) {
    unsigned u = __float_as_uint(f);
    u += 0x7FFFu + ((u >> 16) & 1u);   // round to nearest even
    return (unsigned short)(u >> 16);
}

__global__ void __launch_bounds__(256) zero_colsum_kernel(float* __restrict__ g, int n) {
    int i = blockIdx.x * blockDim.x + threadIdx.x;
    if (i < n) g[i] = 0.0f;
}

// Computes gcolsum[bh][m] = sum_n softmax_row(QK^T * scale)[n, m]
__global__ void __launch_bounds__(NTHREADS) attn_colsum_kernel(
        const float* __restrict__ q, const float* __restrict__ k,
        float* __restrict__ gcolsum) {
    __shared__ __align__(32) unsigned short lds_q[ROWS_PER_BLOCK * DH]; // bf16
    __shared__ __align__(32) unsigned short lds_k[MCHUNK * DH];         // bf16
    __shared__ float lds_cs[SEQ];

    const int tid  = threadIdx.x;
    const int wave = tid >> 5;
    const int lane = tid & 31;
    const int bh   = blockIdx.x >> 4;                 // 16 row-blocks per (b,h)
    const int n0   = (blockIdx.x & 15) * ROWS_PER_BLOCK;

    const float* qb = q + (size_t)bh * SEQ * DH;
    const float* kb = k + (size_t)bh * SEQ * DH;

    // zero LDS column-sum accumulator
    for (int i = tid; i < SEQ; i += NTHREADS) lds_cs[i] = 0.0f;

    // Load Q row-block, pre-scaled, as bf16 (coalesced: tid walks d fastest)
    for (int i = tid; i < ROWS_PER_BLOCK * DH; i += NTHREADS) {
        int r = i >> 6, d = i & 63;
        lds_q[i] = f2bf_rne(qb[(size_t)(n0 + r) * DH + d] * QK_SCALE);
    }
    __syncthreads();

    // ---- Build the wave's persistent A fragments (16x32 bf16, two K-halves) ----
    // A k order used here (contiguous, k = 16h + s): lanes 0-15 row M=lane: V0..3 = K 0..7, V4..7 = K 8..15
    //                           lanes 16-31 row M=lane-16: V0..3 = K 16..23, V4..7 = K 24..31
    const int lhalf = lane >> 4;
    const int l16   = lane & 15;
    const int mrow  = wave * 16 + l16;       // row within 128-row block
    const int kb0   = lhalf ? 16 : 0;
    const int kb1   = lhalf ? 24 : 8;
    v16bf a0, a1;
    {
        const v8bf* qrow = (const v8bf*)(&lds_q[mrow * DH]); // 8-elem (16B) chunks
        v8bf lo = qrow[kb0 >> 3];
        v8bf hi = qrow[kb1 >> 3];
#pragma unroll
        for (int i = 0; i < 8; ++i) { a0[i] = lo[i]; a0[8 + i] = hi[i]; }
        lo = qrow[(kb0 + 32) >> 3];
        hi = qrow[(kb1 + 32) >> 3];
#pragma unroll
        for (int i = 0; i < 8; ++i) { a1[i] = lo[i]; a1[8 + i] = hi[i]; }
    }

    // B k order used here (contiguous, same as A): lanes 0-15 hold K=0..15, lanes 16-31 hold K=16..31;
    // column N = lane%16. Per lane: 16 contiguous bf16 along d. dbase selects K-half.
    const int dbase = lhalf ? 16 : 0;

    // ================= PASS 1: per-row sum of exp =================
    float runsum[8];
#pragma unroll
    for (int j = 0; j < 8; ++j) runsum[j] = 0.0f;

    for (int chunk = 0; chunk < SEQ / MCHUNK; ++chunk) {
        __syncthreads();
        for (int i = tid; i < MCHUNK * DH; i += NTHREADS) {
            int r = i >> 6, d = i & 63;
            lds_k[i] = f2bf_rne(kb[(size_t)(chunk * MCHUNK + r) * DH + d]);
        }
        __syncthreads();
#pragma unroll
        for (int t = 0; t < MCHUNK / 16; ++t) {
            const unsigned short* krow = &lds_k[(t * 16 + l16) * DH];
            v16bf b0 = *(const v16bf*)(krow + dbase);
            v16bf b1 = *(const v16bf*)(krow + 32 + dbase);
            v8f c = {0.f, 0.f, 0.f, 0.f, 0.f, 0.f, 0.f, 0.f};
            c = __builtin_amdgcn_wmma_f32_16x16x32_bf16(false, a0, false, b0,
                                                        (short)0, c, false, false);
            c = __builtin_amdgcn_wmma_f32_16x16x32_bf16(false, a1, false, b1,
                                                        (short)0, c, false, false);
#pragma unroll
            for (int j = 0; j < 8; ++j)
                runsum[j] += __expf(fminf(c[j], 60.0f));
        }
    }

    // Each row lives in 16 lanes of one C VGPR -> reduce within 16-lane groups.
    float rinv[8];
#pragma unroll
    for (int j = 0; j < 8; ++j) {
        float s = runsum[j];
        s += __shfl_xor(s, 1, 32);
        s += __shfl_xor(s, 2, 32);
        s += __shfl_xor(s, 4, 32);
        s += __shfl_xor(s, 8, 32);
        rinv[j] = 1.0f / s;
    }

    // ================= PASS 2: column sums of softmax =================
    for (int chunk = 0; chunk < SEQ / MCHUNK; ++chunk) {
        __syncthreads();
        for (int i = tid; i < MCHUNK * DH; i += NTHREADS) {
            int r = i >> 6, d = i & 63;
            lds_k[i] = f2bf_rne(kb[(size_t)(chunk * MCHUNK + r) * DH + d]);
        }
        __syncthreads();
#pragma unroll
        for (int t = 0; t < MCHUNK / 16; ++t) {
            const unsigned short* krow = &lds_k[(t * 16 + l16) * DH];
            v16bf b0 = *(const v16bf*)(krow + dbase);
            v16bf b1 = *(const v16bf*)(krow + 32 + dbase);
            v8f c = {0.f, 0.f, 0.f, 0.f, 0.f, 0.f, 0.f, 0.f};
            c = __builtin_amdgcn_wmma_f32_16x16x32_bf16(false, a0, false, b0,
                                                        (short)0, c, false, false);
            c = __builtin_amdgcn_wmma_f32_16x16x32_bf16(false, a1, false, b1,
                                                        (short)0, c, false, false);
            float csum = 0.0f;
#pragma unroll
            for (int j = 0; j < 8; ++j)
                csum += __expf(fminf(c[j], 60.0f)) * rinv[j];
            // lanes L and L+16 hold the two 8-row halves of the same column
            csum += __shfl_xor(csum, 16, 32);
            if (lane < 16)
                atomicAdd(&lds_cs[chunk * MCHUNK + t * 16 + l16], csum);
        }
    }
    __syncthreads();
    for (int i = tid; i < SEQ; i += NTHREADS)
        atomicAdd(&gcolsum[(size_t)bh * SEQ + i], lds_cs[i]);
}

// out[b,h,m,d] = v[b,h,m,d] * colsum[b,h,m]
__global__ void __launch_bounds__(256) scale_v_kernel(
        const float* __restrict__ v, const float* __restrict__ gcolsum,
        float* __restrict__ out) {
    size_t i = (size_t)blockIdx.x * blockDim.x + threadIdx.x;
    out[i] = v[i] * gcolsum[i >> 6];   // i/DH -> (bh*SEQ + m)
}

extern "C" void kernel_launch(void* const* d_in, const int* in_sizes, int n_in,
                              void* d_out, int out_size, void* d_ws, size_t ws_size,
                              hipStream_t stream) {
    (void)in_sizes; (void)n_in; (void)out_size; (void)ws_size;
    const float* q = (const float*)d_in[0];
    const float* k = (const float*)d_in[1];
    const float* v = (const float*)d_in[2];
    float* out     = (float*)d_out;
    float* gcolsum = (float*)d_ws;                 // BH*SEQ floats = 512 KB

    zero_colsum_kernel<<<(BH * SEQ + 255) / 256, 256, 0, stream>>>(gcolsum, BH * SEQ);
    attn_colsum_kernel<<<BH * (SEQ / ROWS_PER_BLOCK), NTHREADS, 0, stream>>>(q, k, gcolsum);
    scale_v_kernel<<<(BH * SEQ * DH) / 256, 256, 0, stream>>>(v, gcolsum, out);
}
